// WindowMultiheadSelfAttention_747324309608
// MI455X (gfx1250) — matrix operand order repaired and checked by reading, not compiled, not run on hardware
//
#include <hip/hip_runtime.h>
#include <hip/hip_bf16.h>
#include <stdint.h>

typedef __attribute__((ext_vector_type(16))) __bf16 v16bf;
typedef __attribute__((ext_vector_type(8)))  float  v8f;
typedef unsigned short u16;
typedef unsigned int   u32;
typedef unsigned long long u64;
typedef unsigned int uv4 __attribute__((ext_vector_type(4)));
typedef int          iv4 __attribute__((ext_vector_type(4)));
typedef int          iv8 __attribute__((ext_vector_type(8)));

#define B_TOT   2048
#define SEQ_N   64
#define CH      512
#define NHEAD   16
#define HDIM    32
#define C3      1536
#define NWIN    64
#define SCALE_F 0.17677669529663687f   // 32^-0.5

#if defined(__gfx1250__) && __has_builtin(__builtin_amdgcn_tensor_load_to_lds) && \
    __has_builtin(__builtin_amdgcn_s_wait_tensorcnt)
#define HAVE_TDM 1
#else
#define HAVE_TDM 0
#endif

union FragAB { v16bf v; u16 s[16]; u32 u[8]; };
union FragC  { v8f   v; float f[8]; };

// hardware bf16 convert (RTNE) — expect v_cvt_pk_bf16_f32
__device__ __forceinline__ u16 f2bf(float x) {
    union { __bf16 h; u16 s; } t; t.h = (__bf16)x; return t.s;
}
__device__ __forceinline__ u32 f2bf2(float lo, float hi) {
    union { __bf16 h[2]; u32 u; } t;
    t.h[0] = (__bf16)lo; t.h[1] = (__bf16)hi;
    return t.u;
}

// ---- CDNA5 WMMA fragment loaders (contiguous k = 16*(lane/16) + s, wave32) --------------
// A fragment: 16x32 bf16, LDS row-major T[row][k], row stride STRIDE (u16).
template<int STRIDE>
__device__ __forceinline__ v16bf load_a_frag(const u16* T, int rowBase, int kBase) {
    const int lane = threadIdx.x & 31;
    const u16* row = T + (size_t)(rowBase + (lane & 15)) * STRIDE + kBase;
    const int kh = (lane >> 4) ? 16 : 0;
    FragAB f;
#pragma unroll
    for (int i = 0; i < 8; ++i) {
        const int kk = kh + 2 * i;
        f.u[i] = *(const u32*)(row + kk);
    }
    return f.v;
}
// B fragment: 32x16 bf16 (KxN), LDS holds Bt[n][k] (B column-major).
template<int STRIDE>
__device__ __forceinline__ v16bf load_b_frag(const u16* T, int colBase, int kBase) {
    const int lane = threadIdx.x & 31;
    const u16* p = T + (size_t)(colBase + (lane & 15)) * STRIDE + kBase + ((lane >> 4) << 4);
    FragAB f;
#pragma unroll
    for (int i = 0; i < 8; ++i) f.u[i] = *(const u32*)(p + 2 * i);
    return f.v;
}
// B fragment when LDS holds B row-major T[k][n] — strided gather.
template<int STRIDE>
__device__ __forceinline__ v16bf load_b_frag_rowmajor(const u16* T, int colBase, int kBase) {
    const int lane = threadIdx.x & 31;
    const int n  = colBase + (lane & 15);
    const int kb = kBase + ((lane >> 4) << 4);
    FragAB f;
#pragma unroll
    for (int i = 0; i < 16; ++i) f.s[i] = T[(size_t)(kb + i) * STRIDE + n];
    return f.v;
}

#define WMMA_BF16(A, B, C) \
    __builtin_amdgcn_wmma_f32_16x16x32_bf16(false, (A), false, (B), (short)0, (C), false, false)

#if HAVE_TDM
// One contiguous 6144-u16 (12 KB) region -> LDS with 1-DWORD pad every 64 B,
// reproducing the [3][64][34] padded LDS layout in hardware. D# per ISA §8.
__device__ __forceinline__ void tdm_load_qkv(u32 lds_off, u64 ga) {
    const u32 dim0 = 3 * SEQ_N * HDIM;            // 6144 elements
    uv4 g0;
    g0.x = 1u;                                    // count=1 (valid user descriptor)
    g0.y = lds_off;                               // lds_addr (bytes)
    g0.z = (u32)ga;                               // global_addr[31:0]
    g0.w = (u32)(ga >> 32) | (2u << 30);          // global_addr[56:32] | type=2
    iv8 g1;
    g1[0] = (int)((1u << 16) | (1u << 20) | (3u << 22)); // data_size=2B, pad_enable, pad_interval=64B (pad_amount=1 DWORD)
    g1[1] = (int)((dim0 & 0xFFFFu) << 16);        // tensor_dim0[15:0]
    g1[2] = (int)((dim0 >> 16) | (1u << 16));     // tensor_dim0[31:16] | tensor_dim1=1
    g1[3] = (int)((dim0 & 0xFFFFu) << 16);        // tile_dim0 = 6144
    g1[4] = 0;                                    // tile_dim1/2 unused
    g1[5] = (int)dim0;                            // tensor_dim0_stride
    g1[6] = 0;
    g1[7] = 0;
    iv4 z4 = {0, 0, 0, 0};
#if __clang_major__ >= 23
    iv8 z8 = {0, 0, 0, 0, 0, 0, 0, 0};
    __builtin_amdgcn_tensor_load_to_lds(g0, g1, z4, z4, z8, 0);
#else
    __builtin_amdgcn_tensor_load_to_lds(g0, g1, z4, z4, 0);
#endif
}
#endif

// ---------------------------------------------------------------------------
// Kernel 1: qkv = (x + pe) @ qkv_w^T + qkv_b  -> bf16 ws (b,h,{q,k,v},n,d)
// Block tile 64(M) x 128(N); 8 waves each own a 32x32 quad (2x2 WMMA tiles).
// ---------------------------------------------------------------------------
__global__ __launch_bounds__(256) void qkv_gemm_kernel(
    const float* __restrict__ x, const float* __restrict__ pe,
    const float* __restrict__ qkv_w, const float* __restrict__ qkv_b,
    u16* __restrict__ qkvo)
{
    __shared__ u16 As[64][34];
    __shared__ u16 Bs[128][34];

    const int tid  = threadIdx.x;
    const int wave = tid >> 5;
    const int lane = tid & 31;
    const int row0 = blockIdx.x * 64;
    const int col0 = blockIdx.y * 128;
    const int wm = wave & 1;
    const int wn = wave >> 1;          // 0..3 -> 32-col slices

    FragC acc[2][2] = {};

    const int lr  = tid >> 2;          // 0..63
    const int lk  = (tid & 3) * 8;     // A tile K sub-offset
    const int br  = tid >> 1;          // 0..127
    const int bk  = (tid & 1) * 16;    // B tile K sub-offset

    for (int k0 = 0; k0 < CH; k0 += 32) {
        {   // A tile: (x + pe) -> bf16, 64x32
            const int gr = row0 + lr;
            const float4* xp = (const float4*)(x  + (size_t)gr * CH + k0 + lk);
            const float4* pp = (const float4*)(pe + (size_t)(gr & (SEQ_N - 1)) * CH + k0 + lk);
            const float4 a0 = xp[0], a1 = xp[1];
            const float4 p0 = pp[0], p1 = pp[1];
            u32* dst = (u32*)&As[lr][lk];
            dst[0] = f2bf2(a0.x + p0.x, a0.y + p0.y);
            dst[1] = f2bf2(a0.z + p0.z, a0.w + p0.w);
            dst[2] = f2bf2(a1.x + p1.x, a1.y + p1.y);
            dst[3] = f2bf2(a1.z + p1.z, a1.w + p1.w);
        }
        {   // B tile: qkv_w rows (output cols) -> Bs[j][k], 128x32
            const float4* wp = (const float4*)(qkv_w + (size_t)(col0 + br) * CH + k0 + bk);
            const float4 b0 = wp[0], b1 = wp[1], b2 = wp[2], b3 = wp[3];
            u32* dst = (u32*)&Bs[br][bk];
            dst[0] = f2bf2(b0.x, b0.y); dst[1] = f2bf2(b0.z, b0.w);
            dst[2] = f2bf2(b1.x, b1.y); dst[3] = f2bf2(b1.z, b1.w);
            dst[4] = f2bf2(b2.x, b2.y); dst[5] = f2bf2(b2.z, b2.w);
            dst[6] = f2bf2(b3.x, b3.y); dst[7] = f2bf2(b3.z, b3.w);
        }
        if (k0 + 32 < CH) {   // global_prefetch_b8 of next K tiles
            __builtin_prefetch(x + (size_t)(row0 + lr) * CH + k0 + 32 + lk, 0, 3);
            __builtin_prefetch(qkv_w + (size_t)(col0 + br) * CH + k0 + 32 + bk, 0, 3);
        }
        __syncthreads();

        const v16bf a0f = load_a_frag<34>(&As[0][0], wm * 32, 0);
        const v16bf a1f = load_a_frag<34>(&As[0][0], wm * 32 + 16, 0);
        const v16bf b0f = load_b_frag<34>(&Bs[0][0], wn * 32, 0);
        const v16bf b1f = load_b_frag<34>(&Bs[0][0], wn * 32 + 16, 0);
        acc[0][0].v = WMMA_BF16(a0f, b0f, acc[0][0].v);
        acc[1][0].v = WMMA_BF16(a1f, b0f, acc[1][0].v);
        acc[0][1].v = WMMA_BF16(a0f, b1f, acc[0][1].v);
        acc[1][1].v = WMMA_BF16(a1f, b1f, acc[1][1].v);
        __syncthreads();
    }

    // Scatter to (b, h, {q,k,v}, n, d) bf16 workspace, fused bias.
#pragma unroll
    for (int cn = 0; cn < 2; ++cn) {
        const int j     = col0 + wn * 32 + cn * 16 + (lane & 15);  // 0..1535
        const int which = j >> 9;
        const int h = (j >> 5) & 15, d = j & 31;
        const float bias = qkv_b[j];
#pragma unroll
        for (int sub = 0; sub < 2; ++sub) {
            FragC& c = acc[sub][cn];
#pragma unroll
            for (int r = 0; r < 8; ++r) {
                const int m = row0 + wm * 32 + sub * 16 + r + ((lane >> 4) << 3);
                const int b = m >> 6, n = m & 63;
                qkvo[(((size_t)(b * NHEAD + h) * 3 + which) * SEQ_N + n) * HDIM + d] =
                    f2bf(c.f[r] + bias);
            }
        }
    }
}

// ---------------------------------------------------------------------------
// Kernel 2: per-(b,h) attention. q/k/v staged by Tensor Data Mover.
// ---------------------------------------------------------------------------
__global__ __launch_bounds__(256) void attn_kernel(
    const u16* __restrict__ qkv,
    const float* __restrict__ rpe, const float* __restrict__ mask,
    u16* __restrict__ ao)
{
    __shared__ u16  qkvs[3][64][34];     // q,k,v each 64x32 (stride 34)
    __shared__ float S[64][65];
    __shared__ u16  P[64][66];

    const int tid  = threadIdx.x;
    const int wave = tid >> 5;
    const int lane = tid & 31;
    const int bh = blockIdx.x;
    const int b = bh >> 4, h = bh & 15;
    const int w = b & (NWIN - 1);

    const u16* qs = &qkvs[0][0][0];
    const u16* ks = &qkvs[1][0][0];
    const u16* vs = &qkvs[2][0][0];

#if HAVE_TDM
    if (wave == 0) {
        tdm_load_qkv((u32)(uintptr_t)&qkvs[0][0][0],
                     (u64)(uintptr_t)(qkv + (size_t)bh * (3 * SEQ_N * HDIM)));
        __builtin_amdgcn_s_wait_tensorcnt(0);
    }
#else
    {   // cooperative fallback: 3072 dwords, pad every 16 dwords
        const u32* src = (const u32*)(qkv + (size_t)bh * (3 * SEQ_N * HDIM));
        u32* dst = (u32*)&qkvs[0][0][0];
#pragma unroll
        for (int i = 0; i < 12; ++i) {
            const int gi = tid + i * 256;
            dst[(gi >> 4) * 17 + (gi & 15)] = src[gi];
        }
    }
#endif
    __syncthreads();

    // S = scale * q k^T + rpe[h] + mask[w]   (16 WMMA tiles over 8 waves)
#pragma unroll
    for (int tt = 0; tt < 2; ++tt) {
        const int t  = wave * 2 + tt;
        const int ti = t >> 2, tj = t & 3;
        const v16bf af = load_a_frag<34>(qs, ti * 16, 0);
        const v16bf bf = load_b_frag<34>(ks, tj * 16, 0);   // B[d][keyrow]
        FragC c = {};
        c.v = WMMA_BF16(af, bf, c.v);
        const int n = tj * 16 + (lane & 15);
#pragma unroll
        for (int r = 0; r < 8; ++r) {
            const int m = ti * 16 + r + ((lane >> 4) << 3);
            S[m][n] = c.f[r] * SCALE_F
                    + rpe[((size_t)h * SEQ_N + m) * SEQ_N + n]
                    + mask[((size_t)w * SEQ_N + m) * SEQ_N + n];
        }
    }
    __syncthreads();

    // row softmax
    if (tid < 64) {
        float mx = -3.4e38f;
#pragma unroll 8
        for (int j = 0; j < 64; ++j) mx = fmaxf(mx, S[tid][j]);
        float sum = 0.f;
#pragma unroll 8
        for (int j = 0; j < 64; ++j) { const float e = __expf(S[tid][j] - mx); S[tid][j] = e; sum += e; }
        const float rs = 1.f / sum;
#pragma unroll 8
        for (int j = 0; j < 64; ++j) P[tid][j] = f2bf(S[tid][j] * rs);
    }
    __syncthreads();

    // O = P @ v : 8 tiles (one per wave), K=64 in two WMMA steps
    {
        const int ti = wave >> 1, tj = wave & 1;
        FragC c = {};
#pragma unroll
        for (int kc = 0; kc < 2; ++kc) {
            const v16bf af = load_a_frag<66>(&P[0][0], ti * 16, kc * 32);
            const v16bf bf = load_b_frag_rowmajor<34>(vs, tj * 16, kc * 32); // B[keyrow][d]
            c.v = WMMA_BF16(af, bf, c.v);
        }
        const int d = tj * 16 + (lane & 15);
#pragma unroll
        for (int r = 0; r < 8; ++r) {
            const int m = ti * 16 + r + ((lane >> 4) << 3);
            ao[((size_t)b * SEQ_N + m) * CH + h * HDIM + d] = f2bf(c.f[r]);
        }
    }
}

// ---------------------------------------------------------------------------
// Kernel 3: out = attn_out @ proj_w^T + proj_b  (f32), block tile 64x128
// ---------------------------------------------------------------------------
__global__ __launch_bounds__(256) void proj_gemm_kernel(
    const u16* __restrict__ ain, const float* __restrict__ proj_w,
    const float* __restrict__ proj_b, float* __restrict__ out)
{
    __shared__ u16 As[64][34];
    __shared__ u16 Bs[128][34];

    const int tid  = threadIdx.x;
    const int wave = tid >> 5;
    const int lane = tid & 31;
    const int row0 = blockIdx.x * 64;
    const int col0 = blockIdx.y * 128;
    const int wm = wave & 1, wn = wave >> 1;

    FragC acc[2][2] = {};
    const int lr = tid >> 2, lk = (tid & 3) * 8;
    const int br = tid >> 1, bk = (tid & 1) * 16;

    for (int k0 = 0; k0 < CH; k0 += 32) {
        {   // A tile: already bf16
            const u32* src = (const u32*)(ain + (size_t)(row0 + lr) * CH + k0 + lk);
            u32* dst = (u32*)&As[lr][lk];
#pragma unroll
            for (int i = 0; i < 4; ++i) dst[i] = src[i];
        }
        {   // B tile: proj_w -> bf16, 128x32
            const float4* wp = (const float4*)(proj_w + (size_t)(col0 + br) * CH + k0 + bk);
            const float4 b0 = wp[0], b1 = wp[1], b2 = wp[2], b3 = wp[3];
            u32* dst = (u32*)&Bs[br][bk];
            dst[0] = f2bf2(b0.x, b0.y); dst[1] = f2bf2(b0.z, b0.w);
            dst[2] = f2bf2(b1.x, b1.y); dst[3] = f2bf2(b1.z, b1.w);
            dst[4] = f2bf2(b2.x, b2.y); dst[5] = f2bf2(b2.z, b2.w);
            dst[6] = f2bf2(b3.x, b3.y); dst[7] = f2bf2(b3.z, b3.w);
        }
        if (k0 + 32 < CH) {
            __builtin_prefetch(ain + (size_t)(row0 + lr) * CH + k0 + 32 + lk, 0, 3);
            __builtin_prefetch(proj_w + (size_t)(col0 + br) * CH + k0 + 32 + bk, 0, 3);
        }
        __syncthreads();

        const v16bf a0f = load_a_frag<34>(&As[0][0], wm * 32, 0);
        const v16bf a1f = load_a_frag<34>(&As[0][0], wm * 32 + 16, 0);
        const v16bf b0f = load_b_frag<34>(&Bs[0][0], wn * 32, 0);
        const v16bf b1f = load_b_frag<34>(&Bs[0][0], wn * 32 + 16, 0);
        acc[0][0].v = WMMA_BF16(a0f, b0f, acc[0][0].v);
        acc[1][0].v = WMMA_BF16(a1f, b0f, acc[1][0].v);
        acc[0][1].v = WMMA_BF16(a0f, b1f, acc[0][1].v);
        acc[1][1].v = WMMA_BF16(a1f, b1f, acc[1][1].v);
        __syncthreads();
    }

#pragma unroll
    for (int cn = 0; cn < 2; ++cn) {
        const int j = col0 + wn * 32 + cn * 16 + (lane & 15);
        const float bias = proj_b[j];
#pragma unroll
        for (int sub = 0; sub < 2; ++sub) {
            FragC& c = acc[sub][cn];
#pragma unroll
            for (int r = 0; r < 8; ++r) {
                const int m = row0 + wm * 32 + sub * 16 + r + ((lane >> 4) << 3);
                out[(size_t)m * CH + j] = c.f[r] + bias;
            }
        }
    }
}

// ---------------------------------------------------------------------------
extern "C" void kernel_launch(void* const* d_in, const int* in_sizes, int n_in,
                              void* d_out, int out_size, void* d_ws, size_t ws_size,
                              hipStream_t stream) {
    (void)in_sizes; (void)n_in; (void)out_size; (void)ws_size;
    const float* x      = (const float*)d_in[0];
    const float* pe     = (const float*)d_in[1];
    const float* rpe    = (const float*)d_in[2];
    const float* mask   = (const float*)d_in[3];
    const float* qkv_w  = (const float*)d_in[4];
    const float* qkv_b  = (const float*)d_in[5];
    const float* proj_w = (const float*)d_in[6];
    const float* proj_b = (const float*)d_in[7];
    float* out = (float*)d_out;

    const size_t qkv_elems = (size_t)B_TOT * NHEAD * 3 * SEQ_N * HDIM;  // 201,326,592
    u16* qkvws = (u16*)d_ws;
    u16* aows  = qkvws + qkv_elems;   // (B, N, C) bf16

    qkv_gemm_kernel<<<dim3(B_TOT * SEQ_N / 64, C3 / 128), 256, 0, stream>>>(
        x, pe, qkv_w, qkv_b, qkvws);
    attn_kernel<<<dim3(B_TOT * NHEAD), 256, 0, stream>>>(
        qkvws, rpe, mask, aows);
    proj_gemm_kernel<<<dim3(B_TOT * SEQ_N / 64, CH / 128), 256, 0, stream>>>(
        aows, proj_w, proj_b, out);
}
